// CAWN2_63685775065377
// MI455X (gfx1250) — matrix operand order repaired and checked by reading, not compiled, not run on hardware
//
#include <hip/hip_runtime.h>

// ---------------------------------------------------------------------------
// Fused single-query attention block for MI455X (gfx1250, wave32, WMMA).
// B=4096, NS=1, NN=64, D=256, H=8, DK=32.
//
// Algebraic optimization: with one query row per batch the K/V projections
// are folded through the query / attention weights (low-rank trick), cutting
// FLOPs ~12x; the kernel is HBM-bound on seq (~282MB @ 23.3TB/s ~ 12us).
//
// CDNA5 features used:
//  - v_wmma_f32_16x16x32_bf16 for every GEMM (fp32 accumulate)
//  - global_load_async_to_lds_b128 + s_wait_asynccnt: seq tile for batch bb+1
//    streams into LDS while WMMA phases for batch bb execute (double-duty
//    single fp32 staging buffer: it is drained by the bf16 conversion before
//    the next issue).
//  - A-fragments: ds_load_b128 pairs; B-fragments from transposed bf16
//    weights: contiguous 32B per-lane global loads.
//
// Garbage-row policy: WMMA D[m][n] depends only on A row m / B col n, so
// unused rows/cols (M,N tails beyond TB=8 or HEADS=8) may read in-bounds LDS
// garbage; all stores are guarded. This avoids exec-mask branches per K-step.
// ---------------------------------------------------------------------------

#define DIM     256
#define HEADS   8
#define DKH     32
#define NNB     64
#define TB      8          // batch elements per workgroup
#define NTHR    256        // 8 wave32s
#define LN_EPS  1e-5f

typedef __attribute__((ext_vector_type(16))) __bf16          v16bf;
typedef __attribute__((ext_vector_type(8)))  float           v8f;
typedef __attribute__((ext_vector_type(8)))  unsigned short  u16x8;

union Frag { v16bf v; unsigned short u[16]; u16x8 h[2]; };

__device__ __forceinline__ unsigned short f2bf(float x) {
    union { float f; unsigned u; } c; c.f = x;
    unsigned r = c.u + 0x7FFFu + ((c.u >> 16) & 1u);
    return (unsigned short)(r >> 16);
}

__device__ __forceinline__ v8f wmma_bf16(const Frag& a, const Frag& b, v8f c) {
    return __builtin_amdgcn_wmma_f32_16x16x32_bf16(
        /*neg_a=*/false, a.v, /*neg_b=*/false, b.v,
        /*c_mod=*/(short)0, c, /*reuse_a=*/false, /*reuse_b=*/false);
}

// A fragment (16x32, row-major source, 16-bit): lane holds row (lane&15),
// K = e + 16*half (contiguous [16*half, +16))  -> two 16B contiguous loads.
__device__ __forceinline__ void loadA(Frag& f, const unsigned short* base,
                                      int row, int stride, int k0, int halfe) {
    const unsigned short* p = base + row * stride + k0 + 16 * halfe;
    f.h[0] = *(const u16x8*)(p);
    f.h[1] = *(const u16x8*)(p + 8);
}

// B fragment (32x16, 16-bit) from a TRANSPOSED weight W^T[n][k]: lane needs
// K = e + 16*half for its column -> one contiguous 32B per-lane load.
__device__ __forceinline__ void loadBT(Frag& f, const unsigned short* WT,
                                       int colIdx, int K, int k0, int halfe) {
    const unsigned short* p = WT + (size_t)colIdx * K + k0 + 16 * halfe;
    f.h[0] = *(const u16x8*)(p);
    f.h[1] = *(const u16x8*)(p + 8);
}

// Async 16B global -> LDS copy (CDNA5, tracked by ASYNCcnt).
// lds_off = LDS byte address (dynamic segment assumed to start at 0).
__device__ __forceinline__ void async_ld16(unsigned lds_off, const void* gaddr) {
    asm volatile("global_load_async_to_lds_b128 %0, %1, off"
                 :: "v"(lds_off), "v"((unsigned long long)(size_t)gaddr)
                 : "memory");
}
__device__ __forceinline__ void wait_async0() {
    asm volatile("s_wait_asynccnt 0x0" ::: "memory");
}

// ---------------------------------------------------------------------------
// Weight conversion: fp32 -> bf16, optional transpose ([R][C] -> [C][R]).
// ---------------------------------------------------------------------------
__global__ void cvt_w(const float* __restrict__ in, unsigned short* __restrict__ out,
                      int R, int C, int transpose) {
    int i = blockIdx.x * blockDim.x + threadIdx.x;
    if (i >= R * C) return;
    int r = i / C, c = i % C;
    unsigned short v = f2bf(in[i]);
    if (transpose) out[(size_t)c * R + r] = v;
    else           out[i] = v;
}

// ---------------------------------------------------------------------------
// Fused kernel: one workgroup handles TB=8 batch rows.
// ---------------------------------------------------------------------------
__global__ __launch_bounds__(NTHR)
void fused_attn(const float* __restrict__ src, const float* __restrict__ seq,
                const unsigned char* __restrict__ mask,
                const float* __restrict__ bfc, const float* __restrict__ ln_g,
                const float* __restrict__ ln_b, const float* __restrict__ b1,
                const float* __restrict__ b2,
                const unsigned short* __restrict__ WqT,
                const unsigned short* __restrict__ WkRM,
                const unsigned short* __restrict__ WvT,
                const unsigned short* __restrict__ WfcT,
                const unsigned short* __restrict__ W1T,
                const unsigned short* __restrict__ W2T,
                float* __restrict__ out_z, float* __restrict__ out_attn) {
    extern __shared__ char smem[];
    // fp32 regions
    float* sSeqF  = (float*)smem;                 // [64][256] async staging
    float* sSrcF  = sSeqF + NNB * DIM;            // [TB][256]
    float* sScore = sSrcF + TB * DIM;             // [8][64]
    float* sX     = sScore + HEADS * NNB;         // [TB][256]
    // bf16 (u16) regions
    unsigned short* sSrcH = (unsigned short*)(sX + TB * DIM); // [TB][264]
    unsigned short* sQt   = sSrcH + TB * 264;     // [256][TB]  Q^T
    unsigned short* sR    = sQt + DIM * TB;       // [TB][256][8]
    unsigned short* sSeq  = sR + TB * DIM * HEADS;// [64][264]  bf16 seq
    unsigned short* sAttn = sSeq + NNB * 264;     // [8][64]
    unsigned short* sCt   = sAttn + HEADS * NNB;  // [8][256][TB]
    unsigned short* sAout = sCt + HEADS * DIM * TB; // [TB][256]
    unsigned short* sCat  = sAout + TB * DIM;     // [TB][512]
    unsigned short* sH    = sCat + TB * 2 * DIM;  // [TB][256]

    const int tid   = threadIdx.x;
    const int wave  = tid >> 5;
    const int lane  = tid & 31;
    const int colp  = lane & 15;
    const int halfe = lane >> 4;
    const int b0    = blockIdx.x * TB;
    const unsigned seqF_off = (unsigned)((char*)sSeqF - (char*)smem);

    // ---- kick off async staging of seq for batch 0 -----------------------
    {
        const float* g = seq + ((size_t)b0 * NNB) * DIM;
        for (int i = tid; i < (NNB * DIM) / 4; i += NTHR)
            async_ld16(seqF_off + (unsigned)i * 16u, g + (size_t)i * 4);
    }

    // ---- stage src tile (fp32 + bf16) ------------------------------------
    for (int i = tid; i < TB * DIM; i += NTHR) {
        int bb = i / DIM, n = i % DIM;
        float v = src[(size_t)(b0 + bb) * DIM + n];
        sSrcF[bb * DIM + n] = v;
        sSrcH[bb * 264 + n] = f2bf(v);
    }
    __syncthreads();

    // ---- Q = Src @ Wq  -> Q^T in LDS  (M=TB, K=256, N=256) ---------------
    for (int t = wave; t < 16; t += 8) {
        v8f acc = {0.f,0.f,0.f,0.f,0.f,0.f,0.f,0.f};
        for (int kt = 0; kt < 8; ++kt) {
            Frag a, b;
            loadA(a, sSrcH, colp, 264, kt * 32, halfe);
            loadBT(b, WqT, t * 16 + colp, DIM, kt * 32, halfe);
            acc = wmma_bf16(a, b, acc);
        }
        if (halfe == 0) {
            #pragma unroll
            for (int r = 0; r < TB; ++r)
                sQt[(t * 16 + colp) * TB + r] = f2bf(acc[r]);
        }
    }
    __syncthreads();

    // ---- R[b,h,i] = sum_j Wk[i, h*32+j] * Q[b, h*32+j] -------------------
    // Per head: M=256 (i), K=32, N=TB.  128 WMMA tiles, 16 per wave.
    for (int t = wave; t < HEADS * 16; t += 8) {
        int h = t >> 4, mt = t & 15;
        Frag a, b;
        {   // A: rows of Wk (row-major bf16), K restricted to head h
            const unsigned short* p =
                WkRM + (size_t)(mt * 16 + colp) * DIM + h * DKH + 16 * halfe;
            a.h[0] = *(const u16x8*)(p);
            a.h[1] = *(const u16x8*)(p + 8);
        }
        #pragma unroll
        for (int e = 0; e < 16; ++e) {   // B: Q^T gather (N = batch)
            int k = h * DKH + e + 16 * halfe;
            b.u[e] = sQt[k * TB + colp];
        }
        v8f acc = {0.f,0.f,0.f,0.f,0.f,0.f,0.f,0.f};
        acc = wmma_bf16(a, b, acc);
        if (colp < TB) {
            #pragma unroll
            for (int r = 0; r < 8; ++r) {
                int i = mt * 16 + r + 8 * halfe;
                sR[((colp * DIM) + i) * HEADS + h] = f2bf(acc[r]);
            }
        }
    }
    __syncthreads();

    // ---- per-batch attention (async double-duty staging pipeline) --------
    const float inv_t = 0.17677669529663687f;    // 1/sqrt(32)
    for (int bb = 0; bb < TB; ++bb) {
        const size_t bg = (size_t)(b0 + bb);

        // wait for this batch's async seq tile, then convert fp32 -> bf16
        wait_async0();
        __syncthreads();
        for (int i = tid; i < NNB * DIM; i += NTHR)
            sSeq[(i >> 8) * 264 + (i & 255)] = f2bf(sSeqF[i]);
        __syncthreads();   // sSeqF drained; safe to refill

        // kick off async staging for batch bb+1 (overlaps compute below)
        if (bb + 1 < TB) {
            const float* g = seq + ((bg + 1) * NNB) * (size_t)DIM;
            for (int i = tid; i < (NNB * DIM) / 4; i += NTHR)
                async_ld16(seqF_off + (unsigned)i * 16u, g + (size_t)i * 4);
        }

        // scores = Seq_b (64x256) @ R_b^T (256x8):  4 M-tiles, waves 0..3
        if (wave < 4) {
            int mt = wave;
            v8f acc = {0.f,0.f,0.f,0.f,0.f,0.f,0.f,0.f};
            for (int kt = 0; kt < 8; ++kt) {
                Frag a, b;
                loadA(a, sSeq, mt * 16 + colp, 264, kt * 32, halfe);
                #pragma unroll
                for (int e = 0; e < 16; ++e) {
                    int i = kt * 32 + e + 16 * halfe;
                    b.u[e] = sR[((bb * DIM) + i) * HEADS + colp];
                }
                acc = wmma_bf16(a, b, acc);
            }
            if (colp < HEADS) {
                #pragma unroll
                for (int r = 0; r < 8; ++r) {
                    int n = mt * 16 + r + 8 * halfe;
                    sScore[colp * NNB + n] = acc[r] * inv_t;
                }
            }
        }
        __syncthreads();

        // masked softmax over neighbors; wave w owns head w (2 vals/lane)
        {
            int h = wave;
            int n0 = lane, n1 = lane + 32;
            float s0 = (mask[bg * NNB + n0] != 0) ? -1e10f : sScore[h * NNB + n0];
            float s1 = (mask[bg * NNB + n1] != 0) ? -1e10f : sScore[h * NNB + n1];
            float mx = fmaxf(s0, s1);
            for (int off = 16; off > 0; off >>= 1)
                mx = fmaxf(mx, __shfl_xor(mx, off, 32));
            float e0 = __expf(s0 - mx), e1 = __expf(s1 - mx);
            float sm = e0 + e1;
            for (int off = 16; off > 0; off >>= 1)
                sm += __shfl_xor(sm, off, 32);
            float rs = 1.0f / sm;
            float a0 = e0 * rs, a1 = e1 * rs;
            sAttn[h * NNB + n0] = f2bf(a0);
            sAttn[h * NNB + n1] = f2bf(a1);
            float* am = out_attn + (bg * HEADS + h) * NNB;
            am[n0] = a0; am[n1] = a1;
        }
        __syncthreads();

        // C[h,i] = attn (8x64) @ Seq_b (64x256):  16 N-tiles, 2 K-steps
        for (int t = wave; t < 16; t += 8) {
            v8f acc = {0.f,0.f,0.f,0.f,0.f,0.f,0.f,0.f};
            for (int kt = 0; kt < 2; ++kt) {
                Frag a, b;
                loadA(a, sAttn, colp, NNB, kt * 32, halfe);
                #pragma unroll
                for (int e = 0; e < 16; ++e) {  // B: seq^T gather
                    int n = kt * 32 + e + 16 * halfe;
                    b.u[e] = sSeq[n * 264 + t * 16 + colp];
                }
                acc = wmma_bf16(a, b, acc);
            }
            if (halfe == 0) {
                #pragma unroll
                for (int r = 0; r < HEADS; ++r)
                    sCt[((r * DIM) + t * 16 + colp) * TB + bb] = f2bf(acc[r]);
            }
        }
        __syncthreads();
    }

    // ---- out[b, h*32+j] = Wv_h^T (32x256) @ C_h (256xTB) -----------------
    for (int t = wave; t < HEADS * 2; t += 8) {
        int h = t >> 1, mt = t & 1;
        v8f acc = {0.f,0.f,0.f,0.f,0.f,0.f,0.f,0.f};
        for (int kt = 0; kt < 8; ++kt) {
            Frag a, b;
            {   // A: Wv^T rows (transposed bf16 weight)
                const unsigned short* p =
                    WvT + (size_t)(h * DKH + mt * 16 + colp) * DIM + kt * 32 + 16 * halfe;
                a.h[0] = *(const u16x8*)(p);
                a.h[1] = *(const u16x8*)(p + 8);
            }
            #pragma unroll
            for (int e = 0; e < 16; ++e) {
                int i = kt * 32 + e + 16 * halfe;
                b.u[e] = sCt[((h * DIM) + i) * TB + colp];
            }
            acc = wmma_bf16(a, b, acc);
        }
        if (colp < TB) {
            #pragma unroll
            for (int r = 0; r < 8; ++r) {
                int j = mt * 16 + r + 8 * halfe;
                sAout[colp * DIM + h * DKH + j] = f2bf(acc[r]);
            }
        }
    }
    __syncthreads();

    // ---- fc + bias + residual -> x (fp32) --------------------------------
    for (int t = wave; t < 16; t += 8) {
        v8f acc = {0.f,0.f,0.f,0.f,0.f,0.f,0.f,0.f};
        for (int kt = 0; kt < 8; ++kt) {
            Frag a, b;
            loadA(a, sAout, colp, DIM, kt * 32, halfe);
            loadBT(b, WfcT, t * 16 + colp, DIM, kt * 32, halfe);
            acc = wmma_bf16(a, b, acc);
        }
        if (halfe == 0) {
            int n = t * 16 + colp;
            float bias = bfc[n];
            #pragma unroll
            for (int r = 0; r < TB; ++r)
                sX[r * DIM + n] = acc[r] + bias + sSrcF[r * DIM + n];
        }
    }
    __syncthreads();

    // ---- LayerNorm + concat [x_ln, src] ----------------------------------
    {
        int rb = wave;                 // TB == 8 waves: one row each
        float ss = 0.f, s2 = 0.f, vals[8];
        #pragma unroll
        for (int j = 0; j < 8; ++j) {
            float v = sX[rb * DIM + lane + 32 * j];
            vals[j] = v; ss += v; s2 += v * v;
        }
        for (int off = 16; off > 0; off >>= 1) {
            ss += __shfl_xor(ss, off, 32);
            s2 += __shfl_xor(s2, off, 32);
        }
        float mu   = ss * (1.0f / DIM);
        float var  = s2 * (1.0f / DIM) - mu * mu;
        float rstd = rsqrtf(var + LN_EPS);
        #pragma unroll
        for (int j = 0; j < 8; ++j) {
            int n = lane + 32 * j;
            float xl = (vals[j] - mu) * rstd * ln_g[n] + ln_b[n];
            sCat[rb * 2 * DIM + n]       = f2bf(xl);
            sCat[rb * 2 * DIM + DIM + n] = sSrcH[rb * 264 + n];
        }
    }
    __syncthreads();

    // ---- h = relu(cat (TBx512) @ W1 + b1) --------------------------------
    for (int t = wave; t < 16; t += 8) {
        v8f acc = {0.f,0.f,0.f,0.f,0.f,0.f,0.f,0.f};
        for (int kt = 0; kt < 16; ++kt) {
            Frag a, b;
            loadA(a, sCat, colp, 2 * DIM, kt * 32, halfe);
            loadBT(b, W1T, t * 16 + colp, 2 * DIM, kt * 32, halfe);
            acc = wmma_bf16(a, b, acc);
        }
        if (halfe == 0) {
            int n = t * 16 + colp;
            float bias = b1[n];
            #pragma unroll
            for (int r = 0; r < TB; ++r)
                sH[r * DIM + n] = f2bf(fmaxf(acc[r] + bias, 0.0f));
        }
    }
    __syncthreads();

    // ---- z = h @ W2 + b2 -> global ---------------------------------------
    for (int t = wave; t < 16; t += 8) {
        v8f acc = {0.f,0.f,0.f,0.f,0.f,0.f,0.f,0.f};
        for (int kt = 0; kt < 8; ++kt) {
            Frag a, b;
            loadA(a, sH, colp, DIM, kt * 32, halfe);
            loadBT(b, W2T, t * 16 + colp, DIM, kt * 32, halfe);
            acc = wmma_bf16(a, b, acc);
        }
        if (halfe == 0) {
            int n = t * 16 + colp;
            float bias = b2[n];
            #pragma unroll
            for (int r = 0; r < TB; ++r)
                out_z[((size_t)(b0 + r)) * DIM + n] = acc[r] + bias;
        }
    }
}

// ---------------------------------------------------------------------------
extern "C" void kernel_launch(void* const* d_in, const int* in_sizes, int n_in,
                              void* d_out, int out_size, void* d_ws, size_t ws_size,
                              hipStream_t stream) {
    (void)in_sizes; (void)n_in; (void)out_size; (void)ws_size;
    const float* src  = (const float*)d_in[0];
    const float* seq  = (const float*)d_in[1];
    const unsigned char* mask = (const unsigned char*)d_in[2]; // np.bool_ = 1 byte
    const float* Wq   = (const float*)d_in[3];
    const float* Wk   = (const float*)d_in[4];
    const float* Wv   = (const float*)d_in[5];
    const float* Wfc  = (const float*)d_in[6];
    const float* bfc  = (const float*)d_in[7];
    const float* ln_g = (const float*)d_in[8];
    const float* ln_b = (const float*)d_in[9];
    const float* W1   = (const float*)d_in[10];
    const float* b1   = (const float*)d_in[11];
    const float* W2   = (const float*)d_in[12];
    const float* b2   = (const float*)d_in[13];

    // bf16 weight staging in workspace (total 458752 halfwords = 896 KB)
    unsigned short* ws   = (unsigned short*)d_ws;
    unsigned short* WqT  = ws;                  // W^T [out][in]
    unsigned short* WkRM = WqT  + 65536;        // row-major (used as A)
    unsigned short* WvT  = WkRM + 65536;        // W^T (used as A rows)
    unsigned short* WfcT = WvT  + 65536;
    unsigned short* W1T  = WfcT + 65536;        // [256][512]
    unsigned short* W2T  = W1T  + 131072;

    dim3 blk(256);
    cvt_w<<<dim3((65536  + 255) / 256), blk, 0, stream>>>(Wq,  WqT,  256, 256, 1);
    cvt_w<<<dim3((65536  + 255) / 256), blk, 0, stream>>>(Wk,  WkRM, 256, 256, 0);
    cvt_w<<<dim3((65536  + 255) / 256), blk, 0, stream>>>(Wv,  WvT,  256, 256, 1);
    cvt_w<<<dim3((65536  + 255) / 256), blk, 0, stream>>>(Wfc, WfcT, 256, 256, 1);
    cvt_w<<<dim3((131072 + 255) / 256), blk, 0, stream>>>(W1,  W1T,  512, 256, 1);
    cvt_w<<<dim3((65536  + 255) / 256), blk, 0, stream>>>(W2,  W2T,  256, 256, 1);

    float* out_z    = (float*)d_out;
    float* out_attn = out_z + (size_t)4096 * DIM;

    // dynamic LDS: 20992 floats + 62528 halfwords = 209024 bytes (< 320KB WGP)
    const size_t smem_bytes =
        (size_t)(NNB*DIM + TB*DIM + HEADS*NNB + TB*DIM) * 4 +
        (size_t)(TB*264 + DIM*TB + TB*DIM*HEADS + NNB*264 + HEADS*NNB +
                 HEADS*DIM*TB + TB*DIM + TB*2*DIM + TB*DIM) * 2;

    fused_attn<<<dim3(4096 / TB), dim3(NTHR), smem_bytes, stream>>>(
        src, seq, mask, bfc, ln_g, ln_b, b1, b2,
        WqT, WkRM, WvT, WfcT, W1T, W2T, out_z, out_attn);
}
